// Proposed_45131516346610
// MI455X (gfx1250) — matrix operand order repaired and checked by reading, not compiled, not run on hardware
//
#include <hip/hip_runtime.h>
#include <hip/hip_bf16.h>

// ---------------- types ----------------
typedef __attribute__((ext_vector_type(16))) __bf16          bf16x16;
typedef __attribute__((ext_vector_type(8)))  float           floatx8;
typedef __attribute__((ext_vector_type(16))) unsigned short  ushortx16;
typedef __attribute__((ext_vector_type(8)))  unsigned short  ushortx8;

__device__ __forceinline__ unsigned short f2bf(float f) {
    unsigned int u = __float_as_uint(f);
    u = (u + 0x7FFFu + ((u >> 16) & 1u)) >> 16;   // RNE
    return (unsigned short)u;
}
__device__ __forceinline__ float bf2f(unsigned short h) {
    return __uint_as_float(((unsigned int)h) << 16);
}

// ---------------- weight packer: fp32 -> WMMA A-fragment bf16 layout ----------------
// dst half index = ((mtile*KT + kt)*32 + lane)*16 + h
// A 16x32 bf16 layout (contiguous K, matches B fragment loads): lane M=lane%16,
// element h -> K = 16*(lane/16) + h.
__global__ void pack_w(const float* __restrict__ W, unsigned short* __restrict__ dst,
                       int KT, int Kin, int mode /*0=dense [M][Kin]; 1=conv [M][256][3], K=ktap*256+cin*/) {
    int idx = blockIdx.x * blockDim.x + threadIdx.x;
    int total = 16 * KT * 32 * 16;
    if (idx >= total) return;
    int h    = idx & 15;
    int lane = (idx >> 4) & 31;
    int rest = idx >> 9;
    int kt   = rest % KT;
    int mt   = rest / KT;
    int m    = mt * 16 + (lane & 15);
    int j    = h >> 1;
    int klocal = ((lane >> 4) ? 16 : 0) + h; (void)j;
    int K = kt * 32 + klocal;
    float v = 0.0f;
    if (mode == 1) {
        int cin = K & 255, ktap = K >> 8;
        v = W[(size_t)m * 768 + cin * 3 + ktap];
    } else {
        if (K < Kin) v = W[(size_t)m * Kin + K];
    }
    dst[idx] = f2bf(v);
}

// ---------------- BN fold ----------------
__global__ void fold_bn(const float* __restrict__ cb, const float* __restrict__ g,
                        const float* __restrict__ be, const float* __restrict__ m,
                        const float* __restrict__ v, float* __restrict__ scale,
                        float* __restrict__ shift) {
    int c = threadIdx.x;
    float s = g[c] * rsqrtf(v[c] + 1e-5f);
    scale[c] = s;
    shift[c] = (cb[c] - m[c]) * s + be[c];
}

__global__ void set_ones(float* p) { p[threadIdx.x] = 1.0f; }

// zero the guard rows (l = -1 and l = 1024) of the 4 encoder activation buffers.
__global__ void zero_guards(unsigned short* X, unsigned short* H,
                            unsigned short* T, unsigned short* FEAT) {
    int c = threadIdx.x;
    ptrdiff_t off = blockIdx.x ? (ptrdiff_t)1024 * 256 + c : (ptrdiff_t)-256 + c;
    X[off] = 0; H[off] = 0; T[off] = 0; FEAT[off] = 0;
}

// ---------------- head conv (cin=1, k=3) -> transposed bf16 act [l][256] ----------------
__global__ void head_conv(const float* __restrict__ inp, const float* __restrict__ w,
                          const float* __restrict__ b, unsigned short* __restrict__ X,
                          unsigned short* __restrict__ H) {
    int idx = blockIdx.x * 256 + threadIdx.x;
    int c = idx & 255, l = idx >> 8;
    float a0 = (l > 0)    ? inp[l - 1] : 0.0f;
    float a1 = inp[l];
    float a2 = (l < 1023) ? inp[l + 1] : 0.0f;
    float y = w[c * 3 + 0] * a0 + w[c * 3 + 1] * a1 + w[c * 3 + 2] * a2 + b[c];
    unsigned short hb = f2bf(y);
    X[(size_t)l * 256 + c] = hb;
    H[(size_t)l * 256 + c] = hb;
}

// ---------------- WMMA GEMM: D = W(A-pack, M=256) x Act^T(B) ----------------
// Unified addressing: B row n at Bact + n*Bstride, K-step offset = kt*32 halves.
// (Conv k=3 is this same kernel with Bact pre-shifted by -256 halves into the zero
//  guard row, since ktap*256 + (kt&7)*32 == kt*32.)
// 1 M-tile x 2 N-tiles per wave; software-pipelined fragment loads.
__global__ __launch_bounds__(128)
void gemm16(const unsigned short* __restrict__ Apack, const unsigned short* __restrict__ Bact,
            const float* __restrict__ scale, const float* __restrict__ shift,
            const unsigned short* __restrict__ resid, unsigned short* __restrict__ Out,
            int KT, int Bstride, int relu) {
    const int lane  = threadIdx.x & 31;
    const int wave  = threadIdx.x >> 5;
    const int mtile = blockIdx.y * 4 + wave;
    const int n0    = blockIdx.x * 32;           // two 16-wide N tiles
    const int ncol  = lane & 15;
    const int hi    = lane >> 4;

    floatx8 acc0 = {};
    floatx8 acc1 = {};
    const ushortx16* Ap = (const ushortx16*)Apack + (size_t)mtile * KT * 32 + lane;
    const unsigned short* bp0 = Bact + (size_t)(n0 + ncol) * Bstride + hi * 16;
    const unsigned short* bp1 = bp0 + (size_t)16 * Bstride;

    // prologue: fragments for kt = 0
    ushortx16 au = Ap[0];
    ushortx16 b0 = *(const ushortx16*)bp0;
    ushortx16 b1 = *(const ushortx16*)bp1;

    for (int kt = 0; kt + 1 < KT; ++kt) {
        // issue next K-step's loads before consuming the current fragments
        ushortx16 aun = Ap[(size_t)(kt + 1) * 32];
        ushortx16 b0n = *(const ushortx16*)(bp0 + (size_t)(kt + 1) * 32);
        ushortx16 b1n = *(const ushortx16*)(bp1 + (size_t)(kt + 1) * 32);
        bf16x16 a = __builtin_bit_cast(bf16x16, au);
        acc0 = __builtin_amdgcn_wmma_f32_16x16x32_bf16(false, a, false,
                   __builtin_bit_cast(bf16x16, b0), (short)0, acc0, false, false);
        acc1 = __builtin_amdgcn_wmma_f32_16x16x32_bf16(false, a, false,
                   __builtin_bit_cast(bf16x16, b1), (short)0, acc1, false, false);
        au = aun; b0 = b0n; b1 = b1n;
    }
    {   // epilogue K-step
        bf16x16 a = __builtin_bit_cast(bf16x16, au);
        acc0 = __builtin_amdgcn_wmma_f32_16x16x32_bf16(false, a, false,
                   __builtin_bit_cast(bf16x16, b0), (short)0, acc0, false, false);
        acc1 = __builtin_amdgcn_wmma_f32_16x16x32_bf16(false, a, false,
                   __builtin_bit_cast(bf16x16, b1), (short)0, acc1, false, false);
    }

    const int cbase = mtile * 16 + hi * 8;
#pragma unroll
    for (int t = 0; t < 2; ++t) {
        const int ng = n0 + t * 16 + ncol;
        const floatx8& acc = t ? acc1 : acc0;
        float res[8];
        if (resid) {
            ushortx8 rv = *(const ushortx8*)(resid + (size_t)ng * 256 + cbase);
#pragma unroll
            for (int r = 0; r < 8; ++r) res[r] = bf2f(rv[r]);
        } else {
#pragma unroll
            for (int r = 0; r < 8; ++r) res[r] = 0.0f;
        }
        ushortx8 ov = {};
#pragma unroll
        for (int r = 0; r < 8; ++r) {
            float y = acc[r] * scale[cbase + r] + shift[cbase + r] + res[r];
            if (relu) y = fmaxf(y, 0.0f);
            ov[r] = f2bf(y);
        }
        *(ushortx8*)(Out + (size_t)ng * 256 + cbase) = ov;
    }
}

// ---------------- decoder prep: nearest gather + fc1 ----------------
__global__ void dec_prep(const float* __restrict__ coord, const unsigned short* __restrict__ FEAT,
                         const float* __restrict__ fc1w, const float* __restrict__ fc1b,
                         unsigned short* __restrict__ D0, unsigned short* __restrict__ F1) {
    int qs = blockIdx.x;
    int j  = threadIdx.x;
    int s  = qs >> 14;
    int q  = qs & 16383;
    float vx  = s ? 1.0f : -1.0f;
    const float rxv = 1.0f / 1024.0f;
    float co = coord[q];
    float c  = co + vx * rxv + 1e-6f;
    c = fminf(fmaxf(c, -1.0f + 1e-6f), 1.0f - 1e-6f);
    int idx = (int)rintf((c + 1.0f) * 0.5f * 1023.0f);
    idx = max(0, min(1023, idx));
    float qc  = -1.0f + rxv + 2.0f * rxv * (float)idx;
    float rel = (co - qc) * 1024.0f;
    D0[(size_t)qs * 288 + j] = FEAT[(size_t)idx * 256 + j];
    if (j < 31) D0[(size_t)qs * 288 + 257 + j] = 0;
    float h = fmaxf(fc1w[j] * rel + fc1b[j], 0.0f);
    F1[(size_t)qs * 256 + j] = f2bf(h);
}

// ---------------- 256-wide dot (one wave32 per row) ----------------
__global__ void dot256(const unsigned short* __restrict__ Act, const float* __restrict__ W,
                       const float* __restrict__ bias, float* __restrict__ outv, int NQ) {
    int lane = threadIdx.x & 31;
    int wave = threadIdx.x >> 5;
    int qs   = blockIdx.x * 8 + wave;
    if (qs >= NQ) return;
    ushortx8 v = *(const ushortx8*)(Act + (size_t)qs * 256 + lane * 8);
    float s = 0.0f;
#pragma unroll
    for (int i = 0; i < 8; ++i) s += bf2f(v[i]) * W[lane * 8 + i];
#pragma unroll
    for (int off = 16; off > 0; off >>= 1) s += __shfl_xor(s, off, 32);
    if (lane == 0) outv[qs] = s + bias[0];
}

__global__ void post_fc3(const float* __restrict__ rbuf, unsigned short* __restrict__ D0,
                         float* __restrict__ dist) {
    int qs = blockIdx.x * 256 + threadIdx.x;
    float r = rbuf[qs];
    D0[(size_t)qs * 288 + 256] = f2bf(r);
    dist[qs] = fabsf(r) + 1e-9f;
}

__global__ void combine(const float* __restrict__ pred, const float* __restrict__ dist,
                        float* __restrict__ out) {
    int q = blockIdx.x * 256 + threadIdx.x;
    float d0 = dist[q], d1 = dist[q + 16384];
    float tot = d0 + d1;
    out[q] = pred[q] * (d1 / tot) + pred[q + 16384] * (d0 / tot);
}

// ---------------- host ----------------
extern "C" void kernel_launch(void* const* d_in, const int* in_sizes, int n_in,
                              void* d_out, int out_size, void* d_ws, size_t ws_size,
                              hipStream_t stream) {
    (void)in_sizes; (void)out_size;
    if (n_in < 406) return;

    int p = 0;
    auto F = [&](void) { return (const float*)d_in[p++]; };
    const float* inp    = F();
    const float* coord  = F();
    const float* head_w = F();
    const float* head_b = F();
    struct BlkP { const float *c1w,*c1b,*g1,*be1,*m1,*v1,*c2w,*c2b,*g2,*be2,*m2,*v2; } blk[32];
    for (int b = 0; b < 32; ++b) {
        blk[b].c1w = F(); blk[b].c1b = F();
        blk[b].g1  = F(); blk[b].be1 = F(); blk[b].m1 = F(); blk[b].v1 = F();
        blk[b].c2w = F(); blk[b].c2b = F();
        blk[b].g2  = F(); blk[b].be2 = F(); blk[b].m2 = F(); blk[b].v2 = F();
    }
    const float* tail_w = F();
    const float* tail_b = F();
    const float *iw[5], *ib[5];
    for (int i = 0; i < 5; ++i) { iw[i] = F(); ib[i] = F(); }
    const float* fc1w = F(); const float* fc1b = F();
    const float* fc2w = F(); const float* fc2b = F();
    const float* fc3w = F(); const float* fc3b = F();

    // ---- workspace carve ----
    char* base = (char*)d_ws;
    size_t cur = 0;
    auto carve = [&](size_t bytes) -> char* {
        cur = (cur + 255) & ~(size_t)255;
        char* q = base + cur;
        cur += bytes;
        return q;
    };
    // encoder act buffers: 1026 rows (guard row each side), data ptr = raw + 256
    auto carve_act = [&](void) -> unsigned short* {
        unsigned short* raw = (unsigned short*)carve((size_t)1026 * 256 * 2);
        return raw + 256;
    };
    float* ones   = (float*)carve(256 * 4);
    float* scaleB = (float*)carve(256 * 4);
    float* shiftB = (float*)carve(256 * 4);
    unsigned short* X    = carve_act();
    unsigned short* H    = carve_act();
    unsigned short* T    = carve_act();
    unsigned short* FEAT = carve_act();
    unsigned short* convPack[65];
    for (int i = 0; i < 65; ++i) convPack[i] = (unsigned short*)carve(16 * 24 * 32 * 16 * 2);
    unsigned short* fc2P = (unsigned short*)carve(16 * 8 * 32 * 16 * 2);
    unsigned short* i1P  = (unsigned short*)carve(16 * 9 * 32 * 16 * 2);
    unsigned short* i2P  = (unsigned short*)carve(16 * 8 * 32 * 16 * 2);
    unsigned short* i3P  = (unsigned short*)carve(16 * 8 * 32 * 16 * 2);
    unsigned short* i4P  = (unsigned short*)carve(16 * 8 * 32 * 16 * 2);
    unsigned short* D0   = (unsigned short*)carve((size_t)32768 * 288 * 2);
    unsigned short* B1   = (unsigned short*)carve((size_t)32768 * 256 * 2);
    unsigned short* B2   = (unsigned short*)carve((size_t)32768 * 256 * 2);
    float* rbuf = (float*)carve(32768 * 4);
    float* dist = (float*)carve(32768 * 4);
    float* pred = (float*)carve(32768 * 4);
    if (cur > ws_size) return;

    // ---- pack all GEMM weights ----
    int li = 0;
    for (int b = 0; b < 32; ++b) {
        pack_w<<<768, 256, 0, stream>>>(blk[b].c1w, convPack[li++], 24, 768, 1);
        pack_w<<<768, 256, 0, stream>>>(blk[b].c2w, convPack[li++], 24, 768, 1);
    }
    pack_w<<<768, 256, 0, stream>>>(tail_w, convPack[64], 24, 768, 1);
    pack_w<<<256, 256, 0, stream>>>(fc2w,  fc2P, 8, 256, 0);
    pack_w<<<288, 256, 0, stream>>>(iw[0], i1P,  9, 257, 0);
    pack_w<<<256, 256, 0, stream>>>(iw[1], i2P,  8, 256, 0);
    pack_w<<<256, 256, 0, stream>>>(iw[2], i3P,  8, 256, 0);
    pack_w<<<256, 256, 0, stream>>>(iw[3], i4P,  8, 256, 0);
    set_ones<<<1, 256, 0, stream>>>(ones);
    zero_guards<<<2, 256, 0, stream>>>(X, H, T, FEAT);

    // ---- encoder (conv GEMMs: B base shifted one row into the zero guard) ----
    head_conv<<<1024, 256, 0, stream>>>(inp, head_w, head_b, X, H);
    li = 0;
    for (int b = 0; b < 32; ++b) {
        fold_bn<<<1, 256, 0, stream>>>(blk[b].c1b, blk[b].g1, blk[b].be1, blk[b].m1, blk[b].v1, scaleB, shiftB);
        gemm16<<<dim3(32, 4), 128, 0, stream>>>(convPack[li++], H - 256, scaleB, shiftB,
                                                (const unsigned short*)nullptr, T, 24, 256, 1);
        fold_bn<<<1, 256, 0, stream>>>(blk[b].c2b, blk[b].g2, blk[b].be2, blk[b].m2, blk[b].v2, scaleB, shiftB);
        gemm16<<<dim3(32, 4), 128, 0, stream>>>(convPack[li++], T - 256, scaleB, shiftB,
                                                H, H, 24, 256, 0);
    }
    gemm16<<<dim3(32, 4), 128, 0, stream>>>(convPack[64], H - 256, ones, tail_b, X, FEAT, 24, 256, 0);

    // ---- decoder ----
    dec_prep<<<32768, 256, 0, stream>>>(coord, FEAT, fc1w, fc1b, D0, B1);
    gemm16<<<dim3(1024, 4), 128, 0, stream>>>(fc2P, B1, ones, fc2b,
                                              (const unsigned short*)nullptr, B2, 8, 256, 1);
    dot256<<<4096, 256, 0, stream>>>(B2, fc3w, fc3b, rbuf, 32768);
    post_fc3<<<128, 256, 0, stream>>>(rbuf, D0, dist);
    gemm16<<<dim3(1024, 4), 128, 0, stream>>>(i1P, D0, ones, ib[0],
                                              (const unsigned short*)nullptr, B1, 9, 288, 1);
    gemm16<<<dim3(1024, 4), 128, 0, stream>>>(i2P, B1, ones, ib[1],
                                              (const unsigned short*)nullptr, B2, 8, 256, 1);
    gemm16<<<dim3(1024, 4), 128, 0, stream>>>(i3P, B2, ones, ib[2],
                                              (const unsigned short*)nullptr, B1, 8, 256, 1);
    gemm16<<<dim3(1024, 4), 128, 0, stream>>>(i4P, B1, ones, ib[3],
                                              (const unsigned short*)nullptr, B2, 8, 256, 1);
    dot256<<<4096, 256, 0, stream>>>(B2, iw[4], ib[4], pred, 32768);
    combine<<<64, 256, 0, stream>>>(pred, dist, (float*)d_out);
}
